// SkillPathEncoder_33801392619943
// MI455X (gfx1250) — matrix operand order repaired and checked by reading, not compiled, not run on hardware
//
#include <hip/hip_runtime.h>

// ---------------------------------------------------------------------------
// GCN 2-layer forward for MI455X (gfx1250, wave32).
//   out = GCNConv(relu(GCNConv(x, W1, b1)), W2, b2)
// GEMMs: v_wmma_f32_16x16x32_bf16 with 2-term bf16 error-compensated split
// (hi*hi + hi*lo + lo*hi ~ fp32). All fp32->bf16 conversions hoisted into
// one-shot split kernels; GEMM hot loop is loads + WMMA only.
// Edge aggregation: one edge per wave, scalar (s_load) edge indices,
// float4 gather + native non-returning global_atomic_add_f32 (L2-resident).
// ---------------------------------------------------------------------------

typedef __attribute__((ext_vector_type(16))) __bf16          v16bf;
typedef __attribute__((ext_vector_type(8)))  float           v8f;
typedef __attribute__((ext_vector_type(4)))  unsigned int    v4u;
typedef __attribute__((ext_vector_type(8)))  unsigned int    v8u;

#define D 128   // D_IN == D_HID == D_OUT == 128

__device__ __forceinline__ unsigned short f32_to_bf16_rne(float f) {
  unsigned int u = __builtin_bit_cast(unsigned int, f);
  u += 0x7FFFu + ((u >> 16) & 1u);          // round-to-nearest-even
  return (unsigned short)(u >> 16);
}
__device__ __forceinline__ float bf16_to_f32(unsigned short h) {
  unsigned int u = ((unsigned int)h) << 16;
  return __builtin_bit_cast(float, u);
}
__device__ __forceinline__ void split_bf16(float v, unsigned short& hi, unsigned short& lo) {
  hi = f32_to_bf16_rne(v);
  lo = f32_to_bf16_rne(v - bf16_to_f32(hi));
}

// load A fragment: 8 bf16 at p[0..7] and 8 bf16 at p[16..23] (two b128 loads)
__device__ __forceinline__ v16bf load_a_frag(const unsigned short* p) {
  v4u lo = *(const v4u*)(p);
  v4u hi = *(const v4u*)(p + 16);
  v8u c;
  c[0] = lo[0]; c[1] = lo[1]; c[2] = lo[2]; c[3] = lo[3];
  c[4] = hi[0]; c[5] = hi[1]; c[6] = hi[2]; c[7] = hi[3];
  return __builtin_bit_cast(v16bf, c);
}
// load B fragment: 16 contiguous bf16 (two b128 loads worth, one 32B access)
__device__ __forceinline__ v16bf load_b_frag(const unsigned short* p) {
  return __builtin_bit_cast(v16bf, *(const v8u*)p);
}

// ------------------------- degree / normalization --------------------------

__global__ void gcn_deg_init(float* __restrict__ deg, int n) {
  int i = blockIdx.x * blockDim.x + threadIdx.x;
  if (i < n) deg[i] = 1.0f;                 // self-loop contributes 1
}

__global__ void gcn_deg_count(const long long* __restrict__ dst,
                              float* __restrict__ deg, int E) {
  int e = blockIdx.x * blockDim.x + threadIdx.x;
  if (e < E) {
    (void)__hip_atomic_fetch_add(&deg[dst[e]], 1.0f,
                                 __ATOMIC_RELAXED, __HIP_MEMORY_SCOPE_AGENT);
  }
}

__global__ void gcn_rsqrt(const float* __restrict__ deg,
                          float* __restrict__ dis, int n) {
  int i = blockIdx.x * blockDim.x + threadIdx.x;
  if (i < n) dis[i] = rsqrtf(deg[i]);
}

// --------------------------- one-shot split kernels ------------------------

// X [M][D] fp32 -> Ahi/Alo [M][D] bf16 (row-major, same layout)
__global__ void __launch_bounds__(256)
gcn_split_x(const float* __restrict__ X,
            unsigned short* __restrict__ Ahi, unsigned short* __restrict__ Alo,
            long long n) {
  long long i = (long long)blockIdx.x * blockDim.x + threadIdx.x;
  if (i >= n) return;
  unsigned short h, l;
  split_bf16(X[i], h, l);
  Ahi[i] = h; Alo[i] = l;
}

// W [K][N] fp32 -> Whi/Wlo [N][K] bf16 (transposed so B loads are contiguous)
__global__ void __launch_bounds__(256)
gcn_split_wT(const float* __restrict__ W,
             unsigned short* __restrict__ Whi, unsigned short* __restrict__ Wlo) {
  int i = blockIdx.x * blockDim.x + threadIdx.x;
  if (i >= D * D) return;
  int k = i >> 7, n = i & (D - 1);
  unsigned short h, l;
  split_bf16(W[i], h, l);
  Whi[n * D + k] = h; Wlo[n * D + k] = l;
}

// --------------------- GEMM: HS = rowscale(Xsplit @ Wsplit, dis) -----------
// Block = 256 threads = 8 waves; block covers a 16-row stripe, each wave owns
// a 16-column tile (8*16 = 128 = D). K=128 in 4 chunks of 32, 3 WMMAs each.
// Stores HS (dis[src]-scaled h) and initializes ACC with the same value
// (exactly the self-loop aggregation term).

__global__ void __launch_bounds__(256)
gcn_gemm(const unsigned short* __restrict__ Ahi, const unsigned short* __restrict__ Alo,
         const unsigned short* __restrict__ Whi, const unsigned short* __restrict__ Wlo,
         const float* __restrict__ dis,
         float* __restrict__ HS, float* __restrict__ ACC, int M) {
  const int lane = threadIdx.x & 31;
  const int wave = threadIdx.x >> 5;          // 0..7 -> column tile
  const int row0 = blockIdx.x << 4;           // 16-row stripe
  const int r    = lane & 15;
  const int hl   = lane >> 4;                 // lane half (0 or 1)
  const int n    = (wave << 4) + r;           // output column (B/C lane = col)

  int rr = row0 + r;
  if (rr >= M) rr = M - 1;                    // M % 16 == 0 in practice
  const unsigned short* __restrict__ ah_row = Ahi + (size_t)rr * D;
  const unsigned short* __restrict__ al_row = Alo + (size_t)rr * D;
  const unsigned short* __restrict__ bh_col = Whi + (size_t)n  * D;
  const unsigned short* __restrict__ bl_col = Wlo + (size_t)n  * D;

  v8f acc = {};
  #pragma unroll
  for (int kc = 0; kc < D; kc += 32) {
    // A (16x32 bf16): K = kc + 16*half + [0..15] contiguous
    const int ka = kc + (hl << 4);
    // B (32x16 bf16): lane = col n, K = kc + 16*half + [0..15] contiguous
    const int kb = kc + (hl << 4);
    v16bf ah = load_b_frag(ah_row + ka);
    v16bf al = load_b_frag(al_row + ka);
    v16bf bh = load_b_frag(bh_col + kb);
    v16bf bl = load_b_frag(bl_col + kb);

    // low-order terms first, dominant term last (better summation order)
    acc = __builtin_amdgcn_wmma_f32_16x16x32_bf16(
        false, ah, false, bl, (short)0, acc, false, false);
    acc = __builtin_amdgcn_wmma_f32_16x16x32_bf16(
        false, al, false, bh, (short)0, acc, false, false);
    acc = __builtin_amdgcn_wmma_f32_16x16x32_bf16(
        false, ah, false, bh, (short)0, acc, false, false);
  }

  // C/D layout: lanes 0-15 hold M = v, lanes 16-31 hold M = 8+v (v = VGPR 0..7)
  #pragma unroll
  for (int v = 0; v < 8; ++v) {
    int m = row0 + (hl << 3) + v;
    if (m < M) {
      float val = acc[v] * dis[m];            // pre-scale by dis[src]
      size_t off = (size_t)m * D + n;
      HS[off]  = val;                          // scaled features for gather
      ACC[off] = val;                          // accumulator init = self-loop term
    }
  }
}

// ------------------------- edge scatter aggregation ------------------------
// One edge per wave32: edge index forced wave-uniform (readfirstlane) so
// src/dst come back via scalar loads and address math stays on the SALU.
// Each lane: one float4 gather + 4 non-returning float atomic adds.

__global__ void __launch_bounds__(256)
gcn_edge_agg(const float* __restrict__ hs,
             const long long* __restrict__ src,
             const long long* __restrict__ dst,
             float* __restrict__ acc, int E) {
  int e = blockIdx.x * 8 + (threadIdx.x >> 5);
  e = __builtin_amdgcn_readfirstlane(e);
  if (e >= E) return;
  const int lane = threadIdx.x & 31;
  const int s = (int)src[e];                  // wave-uniform -> s_load
  const int d = (int)dst[e];
  const float4 v = *(const float4*)(hs + (size_t)s * D + lane * 4);
  float* ap = acc + (size_t)d * D + lane * 4;
  (void)__hip_atomic_fetch_add(ap + 0, v.x, __ATOMIC_RELAXED, __HIP_MEMORY_SCOPE_AGENT);
  (void)__hip_atomic_fetch_add(ap + 1, v.y, __ATOMIC_RELAXED, __HIP_MEMORY_SCOPE_AGENT);
  (void)__hip_atomic_fetch_add(ap + 2, v.z, __ATOMIC_RELAXED, __HIP_MEMORY_SCOPE_AGENT);
  (void)__hip_atomic_fetch_add(ap + 3, v.w, __ATOMIC_RELAXED, __HIP_MEMORY_SCOPE_AGENT);
}

// -------- post: dst-scale + bias + relu, emitting bf16 split (layer 1) -----

__global__ void __launch_bounds__(256)
gcn_post_split(const float* __restrict__ acc, const float* __restrict__ dis,
               const float* __restrict__ bias,
               unsigned short* __restrict__ Ahi, unsigned short* __restrict__ Alo,
               int n) {
  int i = blockIdx.x * blockDim.x + threadIdx.x;
  if (i >= n * D) return;
  int m = i >> 7;                              // D == 128
  int f = i & (D - 1);
  float v = fmaxf(acc[i] * dis[m] + bias[f], 0.0f);   // relu
  unsigned short h, l;
  split_bf16(v, h, l);
  Ahi[i] = h; Alo[i] = l;
}

// ------------- post: dst-scale + bias, fp32 output (layer 2) ---------------

__global__ void __launch_bounds__(256)
gcn_post_f32(const float* __restrict__ acc, const float* __restrict__ dis,
             const float* __restrict__ bias, float* __restrict__ out, int n) {
  int i = blockIdx.x * blockDim.x + threadIdx.x;
  if (i >= n * D) return;
  int m = i >> 7;
  int f = i & (D - 1);
  out[i] = acc[i] * dis[m] + bias[f];
}

// ---------------------------------------------------------------------------

static inline size_t align256(size_t x) { return (x + 255) & ~(size_t)255; }

extern "C" void kernel_launch(void* const* d_in, const int* in_sizes, int n_in,
                              void* d_out, int out_size, void* d_ws, size_t ws_size,
                              hipStream_t stream) {
  (void)n_in; (void)out_size; (void)ws_size;
  const float*     x   = (const float*)d_in[0];
  const long long* ei  = (const long long*)d_in[1];   // int64 [2, E]
  const float*     W1  = (const float*)d_in[2];
  const float*     b1  = (const float*)d_in[3];
  const float*     W2  = (const float*)d_in[4];
  const float*     b2  = (const float*)d_in[5];
  float*           out = (float*)d_out;

  const int N = in_sizes[0] / D;
  const int E = in_sizes[1] / 2;
  const long long* src = ei;
  const long long* dst = ei + E;
  const long long  ND  = (long long)N * D;

  // workspace layout (all sub-buffers 256B-aligned)
  char* ws = (char*)d_ws;
  float* deg  = (float*)ws;  ws += align256((size_t)N * 4);
  float* dis  = (float*)ws;  ws += align256((size_t)N * 4);
  float* buf0 = (float*)ws;  ws += align256((size_t)ND * 4);   // HS
  float* buf1 = (float*)ws;  ws += align256((size_t)ND * 4);   // ACC
  unsigned short* Ahi = (unsigned short*)ws;  ws += align256((size_t)ND * 2);
  unsigned short* Alo = (unsigned short*)ws;  ws += align256((size_t)ND * 2);
  unsigned short* Whi = (unsigned short*)ws;  ws += align256((size_t)D * D * 2);
  unsigned short* Wlo = (unsigned short*)ws;

  const int nodeBlocks = (N + 255) / 256;
  const int edgeBlocks = (E + 255) / 256;
  const int featBlocks = (int)((ND + 255) / 256);
  const int wBlocks    = (D * D + 255) / 256;
  const int mTiles     = (N + 15) / 16;
  const int aggBlocks  = (E + 7) / 8;          // one edge per wave, 8 waves/block

  // symmetric normalization (shared by both layers)
  gcn_deg_init <<<nodeBlocks, 256, 0, stream>>>(deg, N);
  gcn_deg_count<<<edgeBlocks, 256, 0, stream>>>(dst, deg, E);
  gcn_rsqrt    <<<nodeBlocks, 256, 0, stream>>>(deg, dis, N);

  // ---- layer 1 ----
  gcn_split_x  <<<featBlocks, 256, 0, stream>>>(x, Ahi, Alo, ND);
  gcn_split_wT <<<wBlocks,    256, 0, stream>>>(W1, Whi, Wlo);
  gcn_gemm     <<<mTiles,     256, 0, stream>>>(Ahi, Alo, Whi, Wlo, dis, buf0, buf1, N);
  gcn_edge_agg <<<aggBlocks,  256, 0, stream>>>(buf0, src, dst, buf1, E);
  gcn_post_split<<<featBlocks,256, 0, stream>>>(buf1, dis, b1, Ahi, Alo, N);

  // ---- layer 2 ----
  gcn_split_wT <<<wBlocks,    256, 0, stream>>>(W2, Whi, Wlo);
  gcn_gemm     <<<mTiles,     256, 0, stream>>>(Ahi, Alo, Whi, Wlo, dis, buf0, buf1, N);
  gcn_edge_agg <<<aggBlocks,  256, 0, stream>>>(buf0, src, dst, buf1, E);
  gcn_post_f32 <<<featBlocks, 256, 0, stream>>>(buf1, dis, b2, out, N);
}
